// HierarchicalTitansMemory_3015067041906
// MI455X (gfx1250) — matrix operand order repaired and checked by reading, not compiled, not run on hardware
//
#include <hip/hip_runtime.h>
#include <stdint.h>

// ---------------------------------------------------------------------------
// Problem constants
// ---------------------------------------------------------------------------
#define NROWS   262144
#define DIM     256
#define HIDDEN  64

// d_out layout (floats)
#define RET_OFF   0
#define LOSS_OFF  67108864
#define SURP_OFF  67108865
#define LT_OFF    67108866
#define NWW_OFF   67108867
#define NMW_OFF   (NWW_OFF + 65536)
#define NWL_OFF   (NMW_OFF + 65536)
#define NML_OFF   (NWL_OFF + 65536)

// ws layout (floats)
#define GW_OFF    0          // grad accumulator W-level (65536)
#define GL_OFF    65536      // grad accumulator L-level (65536)
#define KAGG_OFF  131072     // k_agg (256)
#define SC_OFF    131328     // scalars: 0 blend,1 aW,2 eW,3 tW,4 aL,5 eL,6 tL,7 lt,8 ssum,9 lsum
#define KPART_OFF 131360     // 256 blocks x 256 partial col sums
#define ZERO_TOT  131360     // zero [0, ZERO_TOT)

// LDS layout (bytes), pitches in bf16 elements
#define PW   264             // 256 + 8 pad  (row stride 528 B -> 4-bank skew)
#define PT   72              // 64 + 8 pad   (row stride 144 B -> 36-dword skew)
#define SW_B   0
#define SK_B   135168        // 256*264*2
#define SKT_B  168960        // + 64*264*2
#define SET_B  205824        // + 256*72*2
#define SMEM_B 242688        // + 256*72*2

typedef __attribute__((ext_vector_type(16))) __bf16 v16bf;
typedef __attribute__((ext_vector_type(8)))  float  v8f;

union FragBF {
    v16bf    v;
    uint32_t u[8];
    uint4    q[2];
};

__device__ __forceinline__ unsigned short f2bf(float f) {
    union { float f; uint32_t u; } x; x.f = f;
    uint32_t u = x.u;
    uint32_t r = u + 0x7FFFu + ((u >> 16) & 1u);   // round-to-nearest-even
    return (unsigned short)(r >> 16);
}
__device__ __forceinline__ uint32_t packbf2(float a, float b) {
    return (uint32_t)f2bf(a) | ((uint32_t)f2bf(b) << 16);
}
__device__ __forceinline__ float sigmoidf(float x) {
    return 1.0f / (1.0f + __expf(-x));
}

// ---------------------------------------------------------------------------
// ws zeroing
// ---------------------------------------------------------------------------
__global__ void titans_zero_kernel(float* __restrict__ ws) {
    int i = blockIdx.x * blockDim.x + threadIdx.x;
    if (i < ZERO_TOT) ws[i] = 0.0f;
}

// ---------------------------------------------------------------------------
// k_agg: per-block partial column sums, then final reduce
// ---------------------------------------------------------------------------
__global__ void titans_kagg_partial(const float* __restrict__ k,
                                    float* __restrict__ part) {
    int t = threadIdx.x;                 // column
    long long r0 = (long long)blockIdx.x * 1024;
    float s = 0.0f;
    for (int r = 0; r < 1024; ++r)
        s += k[(r0 + r) * DIM + t];
    part[blockIdx.x * DIM + t] = s;
}

__global__ void titans_kagg_final(const float* __restrict__ part,
                                  float* __restrict__ kagg) {
    int t = threadIdx.x;
    float s = 0.0f;
    for (int b = 0; b < 256; ++b) s += part[b * DIM + t];
    kagg[t] = s * (1.0f / (float)NROWS);
}

// ---------------------------------------------------------------------------
// blend + 6 gate scalars (sigmoid(w . k_agg + b))
// ---------------------------------------------------------------------------
__global__ void titans_scalars_kernel(const float* __restrict__ kagg,
                                      float* __restrict__ sc,
                                      const float* __restrict__ blend_w,
                                      const float* __restrict__ blend_b,
                                      const float* __restrict__ Gw_w,
                                      const float* __restrict__ Gw_b,
                                      const float* __restrict__ Gl_w,
                                      const float* __restrict__ Gl_b) {
    __shared__ float sr[256];
    int t = threadIdx.x;
    float kv = kagg[t];
    for (int i = 0; i < 7; ++i) {
        const float* wp; float bias; int dst;
        if (i == 0)      { wp = blend_w;             bias = blend_b[0];   dst = 0; }
        else if (i < 4)  { wp = Gw_w + (i - 1) * DIM; bias = Gw_b[i - 1]; dst = i; }
        else             { wp = Gl_w + (i - 4) * DIM; bias = Gl_b[i - 4]; dst = i; }
        sr[t] = wp[t] * kv;
        __syncthreads();
        for (int s = 128; s > 0; s >>= 1) {
            if (t < s) sr[t] += sr[t + s];
            __syncthreads();
        }
        if (t == 0) sc[dst] = sigmoidf(sr[0] + bias);
        __syncthreads();
    }
}

// ---------------------------------------------------------------------------
// Main fused kernel (run twice: pass 0 = W-level, pass 1 = L-level).
// Per block: 1024 rows in 16 iterations of 64 rows.
//   pred  = k_tile @ W^T      (bf16 WMMA, f32 acc)  -> retrieved, E = pred - v
//   gradp = E^T @ k_tile      (bf16 WMMA, f32 acc, persistent accumulators)
// ---------------------------------------------------------------------------
__global__ __launch_bounds__(512)
void titans_main_kernel(const float* __restrict__ k,
                        const float* __restrict__ v,
                        const float* __restrict__ W,
                        float* __restrict__ out,
                        float* __restrict__ gradacc,
                        float* __restrict__ statSlot,
                        const float* __restrict__ sc,
                        int pass) {
    extern __shared__ char smem[];
    unsigned short* sW  = (unsigned short*)(smem + SW_B);
    unsigned short* sK  = (unsigned short*)(smem + SK_B);
    unsigned short* sKt = (unsigned short*)(smem + SKT_B);
    unsigned short* sEt = (unsigned short*)(smem + SET_B);
    __shared__ float sred[512];

    const int tid  = threadIdx.x;
    const int lane = tid & 31;
    const int wv   = tid >> 5;          // 0..15
    const int jl   = lane & 15;
    const int hi   = lane >> 4;         // 0 or 1
    const float blend = sc[0];
    const float wsel  = (pass == 0) ? blend : (1.0f - blend);

    // ---- stage W (f32 global -> bf16 LDS, row-major with pad) ----
    for (int c = 0; c < 32; ++c) {
        int fi = c * 2048 + tid * 4;
        float4 w4 = *(const float4*)(W + fi);
        int row = fi >> 8, d = fi & 255;
        uint2 p; p.x = packbf2(w4.x, w4.y); p.y = packbf2(w4.z, w4.w);
        *(uint2*)(&sW[row * PW + d]) = p;
    }

    v8f gacc[16];
#pragma unroll
    for (int t = 0; t < 16; ++t) gacc[t] = (v8f){0,0,0,0,0,0,0,0};
    float stat = 0.0f;

    const long long rowBlockBase = (long long)blockIdx.x * 1024;
    __syncthreads();

    for (int it = 0; it < 16; ++it) {
        const long long rowBase = rowBlockBase + it * 64;

        // ---- stage 64x256 k tile: row-major + transposed bf16 copies ----
#pragma unroll
        for (int c = 0; c < 8; ++c) {
            int fi = c * 2048 + tid * 4;
            int r = fi >> 8, d = fi & 255;
            float4 k4 = *(const float4*)(k + (rowBase + r) * DIM + d);
            uint2 p; p.x = packbf2(k4.x, k4.y); p.y = packbf2(k4.z, k4.w);
            *(uint2*)(&sK[r * PW + d]) = p;
            sKt[(d + 0) * PT + r] = f2bf(k4.x);
            sKt[(d + 1) * PT + r] = f2bf(k4.y);
            sKt[(d + 2) * PT + r] = f2bf(k4.z);
            sKt[(d + 3) * PT + r] = f2bf(k4.w);
        }
        __syncthreads();

        // ---- forward: wave wv owns output columns [wv*16, wv*16+16) ----
        const int ntile = wv;
#pragma unroll
        for (int mt = 0; mt < 4; ++mt) {
            v8f acc = (v8f){0,0,0,0,0,0,0,0};
#pragma unroll
            for (int kk = 0; kk < 8; ++kk) {
                FragBF a, b;
                const unsigned short* ap = &sK[(mt * 16 + jl) * PW + kk * 32 + hi * 8];
                a.q[0] = *(const uint4*)(ap);
                a.q[1] = *(const uint4*)(ap + 16);
                const unsigned short* bp = &sW[(ntile * 16 + jl) * PW + kk * 32 + hi * 8];
                b.q[0] = *(const uint4*)(bp);
                b.q[1] = *(const uint4*)(bp + 16);
                acc = __builtin_amdgcn_wmma_f32_16x16x32_bf16(
                        false, a.v, false, b.v, (short)0, acc, false, false);
            }
            const int col = ntile * 16 + jl;
#pragma unroll
            for (int r = 0; r < 8; r += 2) {
                int n0 = mt * 16 + hi * 8 + r;
                long long gr0 = rowBase + n0;
                float p0 = acc[r], p1 = acc[r + 1];
                float v0 = v[gr0 * DIM + col];
                float v1 = v[(gr0 + 1) * DIM + col];
                float e0 = p0 - v0, e1 = p1 - v1;
                *(uint32_t*)(&sEt[col * PT + n0]) = packbf2(e0, e1);
                float* o0 = out + gr0 * DIM + col;
                float* o1 = o0 + DIM;
                if (pass == 0) {
                    *o0 = wsel * p0;
                    *o1 = wsel * p1;
                    stat += e0 * e0 + e1 * e1;      // surprise = ||wp - v||^2
                } else {
                    float r0 = *o0 + wsel * p0;
                    float r1 = *o1 + wsel * p1;
                    *o0 = r0; *o1 = r1;
                    float d0 = r0 - v0, d1 = r1 - v1;
                    stat += d0 * d0 + d1 * d1;      // loss = ||ret - v||^2
                }
            }
        }
        __syncthreads();

        // ---- gradient: wave wv owns grad rows [wv*16, wv*16+16) ----
#pragma unroll
        for (int kk2 = 0; kk2 < 2; ++kk2) {
            FragBF a;
            const unsigned short* ap = &sEt[(wv * 16 + jl) * PT + kk2 * 32 + hi * 8];
            a.q[0] = *(const uint4*)(ap);
            a.q[1] = *(const uint4*)(ap + 16);
#pragma unroll
            for (int jt = 0; jt < 16; ++jt) {
                FragBF b;
                const unsigned short* bp = &sKt[(jt * 16 + jl) * PT + kk2 * 32 + hi * 8];
                b.q[0] = *(const uint4*)(bp);
                b.q[1] = *(const uint4*)(bp + 16);
                gacc[jt] = __builtin_amdgcn_wmma_f32_16x16x32_bf16(
                             false, a.v, false, b.v, (short)0, gacc[jt], false, false);
            }
        }
        __syncthreads();
    }

    // ---- flush gradient accumulators (one hardware f32 atomic per element) ----
#pragma unroll
    for (int jt = 0; jt < 16; ++jt) {
#pragma unroll
        for (int r = 0; r < 8; ++r) {
            int i = wv * 16 + hi * 8 + r;
            int j = jt * 16 + jl;
            unsafeAtomicAdd(&gradacc[i * DIM + j], gacc[jt][r]);
        }
    }

    // ---- block reduce the scalar stat, one atomic per block ----
    sred[tid] = stat;
    __syncthreads();
    for (int s = 256; s > 0; s >>= 1) {
        if (tid < s) sred[tid] += sred[tid + s];
        __syncthreads();
    }
    if (tid == 0) unsafeAtomicAdd(statSlot, sred[0]);
}

// ---------------------------------------------------------------------------
// surprise / loss finalization + lt-gate MLP (Linear->SiLU->Linear->Sigmoid)
// ---------------------------------------------------------------------------
__global__ void titans_ltgate_kernel(const float* __restrict__ S1_w,
                                     const float* __restrict__ S1_b,
                                     const float* __restrict__ S2_w,
                                     const float* __restrict__ S2_b,
                                     float* __restrict__ ws,
                                     float* __restrict__ out) {
    __shared__ float sr[64];
    int t = threadIdx.x;
    const float* kagg = ws + KAGG_OFF;
    const float inv = 1.0f / 67108864.0f;            // 1/(N*VAL_DIM)
    float surprise = ws[SC_OFF + 8] * inv;
    float loss     = ws[SC_OFF + 9] * inv;
    float a = S1_b[t];
    for (int d = 0; d < DIM; ++d) a += S1_w[t * (DIM + 1) + d] * kagg[d];
    a += S1_w[t * (DIM + 1) + DIM] * surprise;
    float h = a * sigmoidf(a);                       // SiLU
    sr[t] = S2_w[t] * h;
    __syncthreads();
    for (int s = 32; s > 0; s >>= 1) {
        if (t < s) sr[t] += sr[t + s];
        __syncthreads();
    }
    if (t == 0) {
        float lt = sigmoidf(sr[0] + S2_b[0]);
        ws[SC_OFF + 7] = lt;
        out[LT_OFF]   = lt;
        out[SURP_OFF] = surprise;
        out[LOSS_OFF] = loss;
    }
}

// ---------------------------------------------------------------------------
// TTT update: grad clip, momentum, norm-clipped renorms. Single block => exact.
// ---------------------------------------------------------------------------
__global__ __launch_bounds__(1024)
void titans_update_kernel(const float* __restrict__ gacc,
                          const float* __restrict__ W,
                          const float* __restrict__ mom,
                          float* __restrict__ outW,
                          float* __restrict__ outM,
                          const float* __restrict__ sc,
                          int gateBase, int useLT) {
    __shared__ float sr[1024];
    __shared__ float snorm[2];
    int t = threadIdx.x;
    float alpha = sc[gateBase + 0];
    float eta   = sc[gateBase + 1];
    float theta = sc[gateBase + 2];
    if (useLT) {
        float lt = sc[7];
        lt = fminf(fmaxf(lt, 0.0f), 1.0f);
        theta *= lt;
    }
    const float gscale = 2.0f / 67108864.0f;         // 2/(N*val_dim)

    float ssq = 0.0f;
    for (int c = 0; c < 64; ++c) {
        int i = c * 1024 + t;
        float g = gscale * gacc[i];
        g = fminf(fmaxf(g, -0.5f), 0.5f);
        float nm = eta * mom[i] - 0.01f * theta * g;
        outM[i] = nm;
        ssq += nm * nm;
    }
    sr[t] = ssq;
    __syncthreads();
    for (int s = 512; s > 0; s >>= 1) {
        if (t < s) sr[t] += sr[t + s];
        __syncthreads();
    }
    if (t == 0) snorm[0] = sqrtf(sr[0]);
    __syncthreads();
    float nrm  = snorm[0];
    float sc_m = (nrm > 5.0f) ? (5.0f / (nrm + 1e-8f)) : 1.0f;

    float ssq2 = 0.0f;
    for (int c = 0; c < 64; ++c) {
        int i = c * 1024 + t;
        float nm = outM[i] * sc_m;
        outM[i] = nm;
        float nw = (1.0f - alpha) * W[i] + nm;
        outW[i] = nw;
        ssq2 += nw * nw;
    }
    __syncthreads();
    sr[t] = ssq2;
    __syncthreads();
    for (int s = 512; s > 0; s >>= 1) {
        if (t < s) sr[t] += sr[t + s];
        __syncthreads();
    }
    if (t == 0) snorm[1] = sqrtf(sr[0]);
    __syncthreads();
    float n2   = snorm[1];
    float sc_w = (n2 > 10.0f) ? (10.0f / (n2 + 1e-8f)) : 1.0f;
    if (sc_w != 1.0f) {
        for (int c = 0; c < 64; ++c) {
            int i = c * 1024 + t;
            outW[i] *= sc_w;
        }
    }
}

// ---------------------------------------------------------------------------
extern "C" void kernel_launch(void* const* d_in, const int* in_sizes, int n_in,
                              void* d_out, int out_size, void* d_ws, size_t ws_size,
                              hipStream_t stream) {
    const float* k       = (const float*)d_in[0];
    const float* v       = (const float*)d_in[1];
    const float* Ww      = (const float*)d_in[2];
    const float* mom_w   = (const float*)d_in[3];
    const float* Wl      = (const float*)d_in[4];
    const float* mom_l   = (const float*)d_in[5];
    const float* Gw_w    = (const float*)d_in[6];
    const float* Gw_b    = (const float*)d_in[7];
    const float* Gl_w    = (const float*)d_in[8];
    const float* Gl_b    = (const float*)d_in[9];
    const float* S1_w    = (const float*)d_in[10];
    const float* S1_b    = (const float*)d_in[11];
    const float* S2_w    = (const float*)d_in[12];
    const float* S2_b    = (const float*)d_in[13];
    const float* blend_w = (const float*)d_in[14];
    const float* blend_b = (const float*)d_in[15];
    float* out = (float*)d_out;
    float* ws  = (float*)d_ws;

    // allow >64KB dynamic LDS (no-op under compile-only; idempotent, no statics)
    (void)hipFuncSetAttribute((const void*)titans_main_kernel,
                              hipFuncAttributeMaxDynamicSharedMemorySize, SMEM_B);

    titans_zero_kernel<<<(ZERO_TOT + 255) / 256, 256, 0, stream>>>(ws);
    titans_kagg_partial<<<256, 256, 0, stream>>>(k, ws + KPART_OFF);
    titans_kagg_final<<<1, 256, 0, stream>>>(ws + KPART_OFF, ws + KAGG_OFF);
    titans_scalars_kernel<<<1, 256, 0, stream>>>(ws + KAGG_OFF, ws + SC_OFF,
                                                 blend_w, blend_b,
                                                 Gw_w, Gw_b, Gl_w, Gl_b);

    titans_main_kernel<<<256, 512, SMEM_B, stream>>>(k, v, Ww, out,
                                                     ws + GW_OFF, ws + SC_OFF + 8,
                                                     ws + SC_OFF, 0);
    titans_main_kernel<<<256, 512, SMEM_B, stream>>>(k, v, Wl, out,
                                                     ws + GL_OFF, ws + SC_OFF + 9,
                                                     ws + SC_OFF, 1);

    titans_ltgate_kernel<<<1, 64, 0, stream>>>(S1_w, S1_b, S2_w, S2_b, ws, out);

    titans_update_kernel<<<1, 1024, 0, stream>>>(ws + GW_OFF, Ww, mom_w,
                                                 out + NWW_OFF, out + NMW_OFF,
                                                 ws + SC_OFF, 1, 0);
    titans_update_kernel<<<1, 1024, 0, stream>>>(ws + GL_OFF, Wl, mom_l,
                                                 out + NWL_OFF, out + NML_OFF,
                                                 ws + SC_OFF, 4, 1);
}
